// CXLAwareKTransformersExperts_24610162606087
// MI455X (gfx1250) — matrix operand order repaired and checked by reading, not compiled, not run on hardware
//
#include <hip/hip_runtime.h>
#include <hip/hip_bf16.h>
#include <stdint.h>

typedef __attribute__((ext_vector_type(16))) __bf16 v16bf;
typedef __attribute__((ext_vector_type(8)))  float  v8f;

union FragU { v16bf v; uint4 q[2]; };

static constexpr int T_TOK = 8192;   // B*S
static constexpr int DDIM  = 2048;
static constexpr int FDIM  = 4096;
static constexpr int NEXP  = 8;
static constexpr int BM    = 128;    // block tile M
static constexpr int BN    = 128;    // block tile N
static constexpr int BK    = 64;     // LDS stage depth (2 WMMA k-steps)
static constexpr int LDSP  = 72;     // padded LDS row stride (elements, 144B = 16B-aligned)
static constexpr int WTM   = 64;     // wave tile M (4 frags)
static constexpr int WTN   = 64;     // wave tile N (4 frags)

// ---------------- routing ----------------

__global__ void k_zero_hdr(int* counts) {
    if (threadIdx.x < 16) counts[threadIdx.x] = 0;   // counts[8] + counts2[8]
}

__global__ void k_route(const float* __restrict__ logits,
                        int* __restrict__ assign, int* __restrict__ counts) {
    int t = blockIdx.x * blockDim.x + threadIdx.x;
    if (t >= T_TOK) return;
    const float* l = logits + (size_t)t * NEXP;
    int   best = 0;
    float bv   = l[0];
    #pragma unroll
    for (int i = 1; i < NEXP; ++i) {
        float v = l[i];
        if (v > bv) { bv = v; best = i; }   // first-max tie-break like jnp.argmax
    }
    assign[t] = best;
    atomicAdd(&counts[best], 1);
}

__global__ void k_scan8(const int* __restrict__ counts, int* __restrict__ offsets) {
    if (threadIdx.x == 0) {
        int acc = 0;
        #pragma unroll
        for (int e = 0; e < NEXP; ++e) { offsets[e] = acc; acc += counts[e]; }
    }
}

// one block per token: claim a compact row for its expert, gather+convert to bf16
__global__ void k_scatter(const float* __restrict__ X,
                          const int* __restrict__ assign,
                          const int* __restrict__ offsets,
                          int* __restrict__ counts2,
                          int* __restrict__ rowTok,
                          __bf16* __restrict__ Xg) {
    __shared__ int srow;
    int t = blockIdx.x;
    if (threadIdx.x == 0) {
        int e    = assign[t];
        int slot = atomicAdd(&counts2[e], 1);
        int row  = offsets[e] + slot;
        rowTok[row] = t;
        srow = row;
    }
    __syncthreads();
    int row = srow;
    const float* src = X  + (size_t)t   * DDIM;
    __bf16*      dst = Xg + (size_t)row * DDIM;
    for (int i = threadIdx.x; i < DDIM; i += blockDim.x)
        dst[i] = (__bf16)src[i];
}

__global__ void k_cvt_bf16(const float* __restrict__ src, __bf16* __restrict__ dst, size_t n) {
    size_t i      = (size_t)blockIdx.x * blockDim.x + threadIdx.x;
    size_t stride = (size_t)gridDim.x * blockDim.x;
    for (; i < n; i += stride) dst[i] = (__bf16)src[i];
}

// ---------------- grouped GEMM ----------------
// D = A_rows[e-group] @ W[e]^T  (A: [rows,K] bf16, W: [E,N,K] bf16)
// FIRST:  apply SiLU, store bf16 into H[row, n]
// !FIRST: scatter fp32 into Out[rowTok[row], n]
template <bool FIRST>
__global__ __launch_bounds__(128)
void k_moe_gemm(const __bf16* __restrict__ A,
                const __bf16* __restrict__ W,
                __bf16* __restrict__ H,
                float* __restrict__ Out,
                const int* __restrict__ counts,
                const int* __restrict__ offsets,
                const int* __restrict__ rowTok,
                int N, int K, int mtPerExpert) {
    const int e  = blockIdx.y / mtPerExpert;
    const int mt = blockIdx.y % mtPerExpert;
    const int cnt = counts[e];
    if (mt * BM >= cnt) return;

    const int row0   = offsets[e] + mt * BM;
    const int rowEnd = offsets[e] + cnt;
    const int n0     = blockIdx.x * BN;
    const __bf16* Wb = W + (size_t)e * N * K;

    __shared__ __align__(16) __bf16 As[2][BM * LDSP];
    __shared__ __align__(16) __bf16 Bs[2][BN * LDSP];

    const int tid  = threadIdx.x;
    const int lane = tid & 31;
    const int w    = tid >> 5;           // wave 0..3
    const int lm   = lane & 15;
    const int lh   = lane >> 4;
    const int mo   = (w & 1) * WTM;      // wave sub-tile origin in block tile
    const int no   = (w >> 1) * WTN;

    // padding rows (gRow >= rowEnd) just re-read a valid row; the epilogue
    // masks those output rows, and rows of D depend only on their own A row.
    const int aRow = (row0 + tid < rowEnd) ? (row0 + tid) : (rowEnd - 1);
    const __bf16* aBase = A  + (size_t)aRow      * K;
    const __bf16* bBase = Wb + (size_t)(n0 + tid) * K;
    const uint32_t ldsA = (uint32_t)(uintptr_t)&As[0][tid * LDSP];
    const uint32_t ldsB = (uint32_t)(uintptr_t)&Bs[0][tid * LDSP];
    const uint32_t bufStrideA = (uint32_t)(BM * LDSP * sizeof(__bf16));
    const uint32_t bufStrideB = (uint32_t)(BN * LDSP * sizeof(__bf16));

    // async memory->LDS copy of one 128B row chunk-set (8 x b128); the 24-bit
    // instruction offset is added to BOTH the LDS and global addresses.
    auto asyncRow = [](uint32_t lds, const __bf16* g) {
        uint64_t ga = (uint64_t)(uintptr_t)g;
        asm volatile(
            "global_load_async_to_lds_b128 %0, %1, off\n\t"
            "global_load_async_to_lds_b128 %0, %1, off offset:16\n\t"
            "global_load_async_to_lds_b128 %0, %1, off offset:32\n\t"
            "global_load_async_to_lds_b128 %0, %1, off offset:48\n\t"
            "global_load_async_to_lds_b128 %0, %1, off offset:64\n\t"
            "global_load_async_to_lds_b128 %0, %1, off offset:80\n\t"
            "global_load_async_to_lds_b128 %0, %1, off offset:96\n\t"
            "global_load_async_to_lds_b128 %0, %1, off offset:112"
            :: "v"(lds), "v"(ga) : "memory");
    };

    auto loadStage = [&](int buf, int kk) {
        asyncRow(ldsA + (uint32_t)buf * bufStrideA, aBase + kk);
        asyncRow(ldsB + (uint32_t)buf * bufStrideB, bBase + kk);
    };

    auto waitAsync = []() {
        asm volatile("s_wait_asynccnt 0x0" ::: "memory");
    };

    v8f c[4][4] = {};

    // one 16x16x32 k-step across the wave's 4x4 frags
    auto computeK = [&](int buf, int ks) {
        FragU a[4], b[4];
        #pragma unroll
        for (int mi = 0; mi < 4; ++mi) {
            // A layout: lanes 0-15 hold k = ks+[0,8)+[16,24); lanes 16-31 k = ks+[8,16)+[24,32)
            const __bf16* pa = &As[buf][(mo + mi * 16 + lm) * LDSP + ks + lh * 8];
            a[mi].q[0] = *reinterpret_cast<const uint4*>(pa);
            a[mi].q[1] = *reinterpret_cast<const uint4*>(pa + 16);
        }
        #pragma unroll
        for (int ni = 0; ni < 4; ++ni) {
            // B layout: lanes 0-15 hold k = ks+[0,8)+[16,24); lanes 16-31 k = ks+[8,16)+[24,32)
            const __bf16* pb = &Bs[buf][(no + ni * 16 + lm) * LDSP + ks + lh * 8];
            b[ni].q[0] = *reinterpret_cast<const uint4*>(pb);
            b[ni].q[1] = *reinterpret_cast<const uint4*>(pb + 16);
        }
        #pragma unroll
        for (int mi = 0; mi < 4; ++mi)
            #pragma unroll
            for (int ni = 0; ni < 4; ++ni)
                c[mi][ni] = __builtin_amdgcn_wmma_f32_16x16x32_bf16(
                    false, a[mi].v, false, b[ni].v, (short)0, c[mi][ni], false, false);
    };

    loadStage(0, 0);
    waitAsync();
    __syncthreads();
    for (int kk = 0; kk < K; kk += BK) {
        const int buf = (kk >> 6) & 1;
        if (kk + BK < K) loadStage(buf ^ 1, kk + BK);
        computeK(buf, 0);
        computeK(buf, 32);
        waitAsync();          // next stage's async writes have landed
        __syncthreads();
    }

    // epilogue: C/D layout -> m = r + 8*lh, n = lm
    #pragma unroll
    for (int mi = 0; mi < 4; ++mi) {
        #pragma unroll
        for (int ni = 0; ni < 4; ++ni) {
            const int gn = n0 + no + ni * 16 + lm;
            #pragma unroll
            for (int r = 0; r < 8; ++r) {
                const int gm = row0 + mo + mi * 16 + 8 * lh + r;
                if (gm >= rowEnd) continue;
                const float x = c[mi][ni][r];
                if (FIRST) {
                    const float s = x / (1.0f + __expf(-x));   // SiLU
                    H[(size_t)gm * N + gn] = (__bf16)s;
                } else {
                    const int tok = rowTok[gm];
                    Out[(size_t)tok * N + gn] = x;
                }
            }
        }
    }
}

// ---------------- launcher ----------------

extern "C" void kernel_launch(void* const* d_in, const int* in_sizes, int n_in,
                              void* d_out, int out_size, void* d_ws, size_t ws_size,
                              hipStream_t stream) {
    (void)in_sizes; (void)n_in; (void)out_size; (void)ws_size;
    const float* X  = (const float*)d_in[0];   // [B,S,D]
    const float* RL = (const float*)d_in[1];   // [B,S,E]
    const float* W1 = (const float*)d_in[2];   // [E,F,D]
    const float* W2 = (const float*)d_in[3];   // [E,D,F]
    float* Out = (float*)d_out;

    constexpr int MT = T_TOK / BM;             // m-tiles per expert (upper bound) = 64

    char* ws = (char*)d_ws;
    int* counts  = (int*)ws;                   // 8
    int* counts2 = counts + 8;                 // 8
    int* offsets = counts + 16;                // 8
    int* assign  = (int*)(ws + 256);           // T
    int* rowTok  = assign + T_TOK;             // T
    __bf16* Xg   = (__bf16*)(ws + 256 + 2 * sizeof(int) * T_TOK);
    __bf16* Hbuf = Xg   + (size_t)T_TOK * DDIM;
    __bf16* W1b  = Hbuf + (size_t)T_TOK * FDIM;
    __bf16* W2b  = W1b  + (size_t)NEXP * FDIM * DDIM;

    k_zero_hdr<<<1, 32, 0, stream>>>(counts);
    k_route<<<T_TOK / 256, 256, 0, stream>>>(RL, assign, counts);
    k_scan8<<<1, 32, 0, stream>>>(counts, offsets);
    k_scatter<<<T_TOK, 256, 0, stream>>>(X, assign, offsets, counts2, rowTok, Xg);
    k_cvt_bf16<<<2048, 256, 0, stream>>>(W1, W1b, (size_t)NEXP * FDIM * DDIM);
    k_cvt_bf16<<<2048, 256, 0, stream>>>(W2, W2b, (size_t)NEXP * DDIM * FDIM);

    // GEMM1: H = silu(Xg @ W1^T)   [rows x F], K = D
    k_moe_gemm<true><<<dim3(FDIM / BN, NEXP * MT), 128, 0, stream>>>(
        Xg, W1b, Hbuf, nullptr, counts, offsets, rowTok, FDIM, DDIM, MT);
    // GEMM2: Out[tok] = H @ W2^T   [rows x D], K = F, scattered by rowTok
    k_moe_gemm<false><<<dim3(DDIM / BN, NEXP * MT), 128, 0, stream>>>(
        Hbuf, W2b, nullptr, Out, counts, offsets, rowTok, DDIM, FDIM, MT);
}
